// MALNet_50989851738865
// MI455X (gfx1250) — compile-verified, matrix operand order checked by reading, not run on hardware
//
#include <hip/hip_runtime.h>
#include <math.h>

// ============================================================================
// MALNet block forward for gfx1250 (MI455X), compile-only target.
//
// d_in: index 0 = x (4,128,60,60) f32, then params flattened with
// alphabetically-sorted keys at every nesting level (jax pytree order).
//
// Dense linears -> bf16 WMMA GEMM (v_wmma_f32_16x16x32_bf16):
//   - 256-thread block = 8 waves (wave32); 4x2 wave grid on a 128x64 C tile;
//     each wave owns 32x32 -> 4 accumulators, 4 WMMA per K-step
//   - double-buffered LDS + software pipelining: global loads for tile k+1
//     issue before the WMMAs of tile k, conversion+ds_store after, single
//     barrier per K-step
//   - f32 -> bf16 pair packing via v_cvt_pk_bf16_f32
//   - edge blocks: clamped addresses + mask-multiply so staging loads are
//     unconditional (no exec-mask predication ladders)
//   - LDS rows padded to 40 bf16 (80B) so every fragment half is one
//     16B-aligned ds_load_b128
// Mamba selective scan fused in registers; 16-state contraction via 16-wide
// __shfl_xor (one channel per half-warp), avoiding dA/dBx materialization.
// ============================================================================

typedef __bf16 bf16;
typedef __attribute__((ext_vector_type(16))) __bf16 v16bf;
typedef __attribute__((ext_vector_type(8)))  float  v8f;

#define DEV __device__ __forceinline__

DEV float sigm(float x)     { return 1.f / (1.f + __expf(-x)); }
DEV float siluf(float x)    { return x * sigm(x); }
DEV float geluf(float x)    { return 0.5f * x * (1.f + erff(x * 0.70710678118f)); }
DEV float hswishf(float x)  { float t = fminf(fmaxf(x + 3.f, 0.f), 6.f); return x * t * (1.f/6.f); }
DEV float softplusf(float x){ return (x > 20.f) ? x : log1pf(__expf(x)); }

// act modes: 0 none, 1 relu, 2 gelu, 3 silu, 4 sigmoid, 5 hswish
DEV float act_apply(float v, int m) {
  switch (m) {
    case 1: return fmaxf(v, 0.f);
    case 2: return geluf(v);
    case 3: return siluf(v);
    case 4: return sigm(v);
    case 5: return hswishf(v);
    default: return v;
  }
}

// pack two f32 into packed bf16 pair (v_cvt_pk_bf16_f32 when available)
DEV unsigned int pack2_bf16(float a, float b) {
#if __has_builtin(__builtin_amdgcn_cvt_pk_bf16_f32)
  auto r = __builtin_amdgcn_cvt_pk_bf16_f32(a, b);
  union U { decltype(r) v; unsigned int u; } c;
  c.v = r;
  return c.u;
#else
  union U { __bf16 h[2]; unsigned int u; } c;
  c.h[0] = (__bf16)a; c.h[1] = (__bf16)b;
  return c.u;
#endif
}

// ---------------------------------------------------------------------------
// WMMA GEMM:  C[M,N] = act(A[M,K] @ W[N,K]^T + bias[N]) * postmul[N]
// ---------------------------------------------------------------------------
#define BM 128
#define BN 64
#define BK 32
#define LDK 40   // padded LDS row (bf16): 80B rows keep 16B fragment chunks aligned

union Frag { v16bf v; uint4 q[2]; };

__global__ __launch_bounds__(256)
void k_gemm_wmma(const float* __restrict__ A, const float* __restrict__ W,
                 const float* __restrict__ bias, const float* __restrict__ postmul,
                 float* __restrict__ C, int M, int N, int K, int act)
{
  __shared__ bf16 As[2][BM][LDK];
  __shared__ bf16 Bs[2][BN][LDK];

  const int t    = threadIdx.x;
  const int lane = t & 31;
  const int wid  = t >> 5;        // 8 waves
  const int wm   = wid & 3;       // wave M subtile (4 x 32 rows)
  const int wn   = wid >> 2;      // wave N subtile (2 x 32 cols)
  const int tileM = blockIdx.y * BM;
  const int tileN = blockIdx.x * BN;

  // cooperative staging maps
  const int lrA = t >> 1;          // 0..127
  const int lcA = (t & 1) * 16;    // 0 or 16
  const int lrB = t >> 2;          // 0..63
  const int lcB = (t & 3) * 8;     // 0,8,16,24

  v8f acc00 = {}, acc01 = {}, acc10 = {}, acc11 = {};

  const int hr = lane & 15;
  const int kg = lane >> 4;

  // ---- one MMA step on LDS buffer `bf_` ----
  auto mma_step = [&](int bf_) {
    Frag fa0, fa1, fb0, fb1;
    fa0.q[0] = *(const uint4*)&As[bf_][wm * 32 + hr][kg * 8];
    fa0.q[1] = *(const uint4*)&As[bf_][wm * 32 + hr][16 + kg * 8];
    fa1.q[0] = *(const uint4*)&As[bf_][wm * 32 + 16 + hr][kg * 8];
    fa1.q[1] = *(const uint4*)&As[bf_][wm * 32 + 16 + hr][16 + kg * 8];
    fb0.q[0] = *(const uint4*)&Bs[bf_][wn * 32 + hr][kg * 8];
    fb0.q[1] = *(const uint4*)&Bs[bf_][wn * 32 + hr][16 + kg * 8];
    fb1.q[0] = *(const uint4*)&Bs[bf_][wn * 32 + 16 + hr][kg * 8];
    fb1.q[1] = *(const uint4*)&Bs[bf_][wn * 32 + 16 + hr][16 + kg * 8];
    acc00 = __builtin_amdgcn_wmma_f32_16x16x32_bf16(false, fa0.v, false, fb0.v,
                                                    (short)0, acc00, false, false);
    acc01 = __builtin_amdgcn_wmma_f32_16x16x32_bf16(false, fa0.v, false, fb1.v,
                                                    (short)0, acc01, false, false);
    acc10 = __builtin_amdgcn_wmma_f32_16x16x32_bf16(false, fa1.v, false, fb0.v,
                                                    (short)0, acc10, false, false);
    acc11 = __builtin_amdgcn_wmma_f32_16x16x32_bf16(false, fa1.v, false, fb1.v,
                                                    (short)0, acc11, false, false);
  };

  // ---- convert+store staged registers into LDS buffer `bf_` ----
  auto store_stageA = [&](int bf_, const float4 f[4]) {
    uint4 q0, q1;
    q0.x = pack2_bf16(f[0].x, f[0].y); q0.y = pack2_bf16(f[0].z, f[0].w);
    q0.z = pack2_bf16(f[1].x, f[1].y); q0.w = pack2_bf16(f[1].z, f[1].w);
    q1.x = pack2_bf16(f[2].x, f[2].y); q1.y = pack2_bf16(f[2].z, f[2].w);
    q1.z = pack2_bf16(f[3].x, f[3].y); q1.w = pack2_bf16(f[3].z, f[3].w);
    *(uint4*)&As[bf_][lrA][lcA]     = q0;
    *(uint4*)&As[bf_][lrA][lcA + 8] = q1;
  };
  auto store_stageB = [&](int bf_, const float4 f[2]) {
    uint4 q;
    q.x = pack2_bf16(f[0].x, f[0].y); q.y = pack2_bf16(f[0].z, f[0].w);
    q.z = pack2_bf16(f[1].x, f[1].y); q.w = pack2_bf16(f[1].z, f[1].w);
    *(uint4*)&Bs[bf_][lrB][lcB] = q;
  };

  const bool fast = ((K % BK) == 0) && (tileM + BM <= M) && (tileN + BN <= N);

  if (fast) {
    // -------- pipelined fast path (interior blocks) --------
    const float* rowA = A + (size_t)(tileM + lrA) * K + lcA;
    const float* rowB = W + (size_t)(tileN + lrB) * K + lcB;

    {  // prologue: stage k0 = 0 into buffer 0
      float4 fa[4], fb[2];
      const float4* sa = (const float4*)rowA;
      fa[0] = sa[0]; fa[1] = sa[1]; fa[2] = sa[2]; fa[3] = sa[3];
      const float4* sb = (const float4*)rowB;
      fb[0] = sb[0]; fb[1] = sb[1];
      store_stageA(0, fa);
      store_stageB(0, fb);
    }
    __syncthreads();

    int buf = 0;
    for (int k0 = 0; k0 < K; k0 += BK) {
      const int kn = k0 + BK;
      const bool more = kn < K;
      float4 fa[4], fb[2];
      if (more) {      // issue next-tile global loads before this tile's WMMAs
        const float4* sa = (const float4*)(rowA + kn);
        fa[0] = sa[0]; fa[1] = sa[1]; fa[2] = sa[2]; fa[3] = sa[3];
        const float4* sb = (const float4*)(rowB + kn);
        fb[0] = sb[0]; fb[1] = sb[1];
        if (kn + BK < K) {   // prefetch tile k+2 toward L2 (global_prefetch_b8)
          __builtin_prefetch(rowA + kn + BK, 0, 1);
          __builtin_prefetch(rowB + kn + BK, 0, 1);
        }
      }
      mma_step(buf);   // ds_load_b128 x8 + 4x v_wmma, overlapping the loads above
      if (more) {
        store_stageA(buf ^ 1, fa);
        store_stageB(buf ^ 1, fb);
      }
      __syncthreads();
      buf ^= 1;
    }
  } else {
    // -------- general path (edge blocks) --------
    // Clamped addresses keep every load legal; mask-MULTIPLY (not select)
    // forces the loads to stay unconditional -- v*0.f is not foldable without
    // fast-math, so the compiler cannot re-predicate the loads.
    for (int k0 = 0; k0 < K; k0 += BK) {
      {
        const int m  = tileM + lrA;
        const int mc = min(m, M - 1);
        const float rmask = (m < M) ? 1.f : 0.f;
        float f[16];
#pragma unroll
        for (int j = 0; j < 16; ++j) {
          const int k  = k0 + lcA + j;
          const int kc = min(k, K - 1);
          const float msk = (k < K) ? rmask : 0.f;
          f[j] = A[(size_t)mc * K + kc] * msk;     // unconditional load
        }
        uint4 q0, q1;
        q0.x = pack2_bf16(f[0], f[1]);   q0.y = pack2_bf16(f[2], f[3]);
        q0.z = pack2_bf16(f[4], f[5]);   q0.w = pack2_bf16(f[6], f[7]);
        q1.x = pack2_bf16(f[8], f[9]);   q1.y = pack2_bf16(f[10], f[11]);
        q1.z = pack2_bf16(f[12], f[13]); q1.w = pack2_bf16(f[14], f[15]);
        *(uint4*)&As[0][lrA][lcA]     = q0;
        *(uint4*)&As[0][lrA][lcA + 8] = q1;
      }
      {
        const int n  = tileN + lrB;
        const int nc = min(n, N - 1);
        const float rmask = (n < N) ? 1.f : 0.f;
        float f[8];
#pragma unroll
        for (int j = 0; j < 8; ++j) {
          const int k  = k0 + lcB + j;
          const int kc = min(k, K - 1);
          const float msk = (k < K) ? rmask : 0.f;
          f[j] = W[(size_t)nc * K + kc] * msk;     // unconditional load
        }
        uint4 q;
        q.x = pack2_bf16(f[0], f[1]); q.y = pack2_bf16(f[2], f[3]);
        q.z = pack2_bf16(f[4], f[5]); q.w = pack2_bf16(f[6], f[7]);
        *(uint4*)&Bs[0][lrB][lcB] = q;
      }
      __syncthreads();
      mma_step(0);
      __syncthreads();
    }
  }

  // ---- epilogue: C/D layout: VGPR r -> row = r + (lane>>4)*8, col = lane&15
#pragma unroll
  for (int i = 0; i < 2; ++i) {
    const v8f* accs[2] = { i == 0 ? &acc00 : &acc10, i == 0 ? &acc01 : &acc11 };
#pragma unroll
    for (int r = 0; r < 8; ++r) {
      const int m = tileM + wm * 32 + i * 16 + (lane >> 4) * 8 + r;
      if (m >= M) continue;
#pragma unroll
      for (int j = 0; j < 2; ++j) {
        const int n = tileN + wn * 32 + j * 16 + hr;
        if (n < N) {
          float v = (*accs[j])[r];
          if (bias)    v += bias[n];
          v = act_apply(v, act);
          if (postmul) v *= postmul[n];
          C[(size_t)m * N + n] = v;
        }
      }
    }
  }
}

// ---------------------------------------------------------------------------
// LayerNorm over last dim of [M, C]
// ---------------------------------------------------------------------------
__global__ void k_ln(const float* __restrict__ x, const float* __restrict__ w,
                     const float* __restrict__ b, float* __restrict__ y,
                     int M, int C, float eps)
{
  __shared__ float red[128];
  const int row = blockIdx.x;
  if (row >= M) return;
  const float* xr = x + (size_t)row * C;

  float s = 0.f;
  for (int c = threadIdx.x; c < C; c += blockDim.x) s += xr[c];
  red[threadIdx.x] = s; __syncthreads();
  for (int o = blockDim.x >> 1; o > 0; o >>= 1) {
    if ((int)threadIdx.x < o) red[threadIdx.x] += red[threadIdx.x + o];
    __syncthreads();
  }
  const float mean = red[0] / C; __syncthreads();

  float v = 0.f;
  for (int c = threadIdx.x; c < C; c += blockDim.x) { float d = xr[c] - mean; v += d * d; }
  red[threadIdx.x] = v; __syncthreads();
  for (int o = blockDim.x >> 1; o > 0; o >>= 1) {
    if ((int)threadIdx.x < o) red[threadIdx.x] += red[threadIdx.x + o];
    __syncthreads();
  }
  const float inv = rsqrtf(red[0] / C + eps);

  for (int c = threadIdx.x; c < C; c += blockDim.x)
    y[(size_t)row * C + c] = (xr[c] - mean) * inv * w[c] + b[c];
}

// ---------------------------------------------------------------------------
// Causal depthwise conv1d over L (D_CONV=4) + SiLU
// ---------------------------------------------------------------------------
__global__ void k_conv1d_silu(const float* __restrict__ xz, int ld, int off,
                              const float* __restrict__ w, const float* __restrict__ bias,
                              float* __restrict__ xs, int B, int L, int di)
{
  const long i = (long)blockIdx.x * blockDim.x + threadIdx.x;
  const long tot = (long)B * L * di;
  if (i >= tot) return;
  const int d = (int)(i % di);
  const long bl = i / di;
  const int l = (int)(bl % L);
  const int b = (int)(bl / L);
  float acc = bias[d];
#pragma unroll
  for (int j = 0; j < 4; ++j) {
    const int ll = l - 3 + j;
    if (ll >= 0) acc += w[d * 4 + j] * xz[(size_t)(b * L + ll) * ld + off + d];
  }
  xs[i] = siluf(acc);
}

// dt[m,d] = softplus(sum_r dbc[m,r] * dt_w[d,r] + dt_b[d])
__global__ void k_dt_softplus(const float* __restrict__ dbc, int ldd,
                              const float* __restrict__ dt_w, const float* __restrict__ dt_b,
                              float* __restrict__ dt, int M, int di, int dr)
{
  const long i = (long)blockIdx.x * blockDim.x + threadIdx.x;
  if (i >= (long)M * di) return;
  const int d = (int)(i % di);
  const int m = (int)(i / di);
  float acc = dt_b[d];
  for (int r = 0; r < dr; ++r) acc += dbc[(size_t)m * ldd + r] * dt_w[d * dr + r];
  dt[i] = softplusf(acc);
}

// ---------------------------------------------------------------------------
// Fused selective scan: one 16-lane half-warp per channel, sequential over L
// ---------------------------------------------------------------------------
__global__ __launch_bounds__(256)
void k_scan(const float* __restrict__ dt, const float* __restrict__ xs,
            const float* __restrict__ dbc, int ldd, int dr,
            const float* __restrict__ zbase, int ldz,
            const float* __restrict__ A_log, const float* __restrict__ Dp,
            float* __restrict__ out, int B, int L, int di)
{
  const int hw = threadIdx.x >> 4;       // 0..15 channel slot
  const int s  = threadIdx.x & 15;       // state index
  const int gpb = di >> 4;               // channel groups per batch
  const int b  = blockIdx.x / gpb;
  const int d  = (blockIdx.x % gpb) * 16 + hw;

  const float A  = -__expf(A_log[d * 16 + s]);
  const float Dd = Dp[d];
  float h = 0.f;

  for (int l = 0; l < L; ++l) {
    const size_t m = (size_t)b * L + l;
    const float dtv = dt[m * di + d];
    const float xv  = xs[m * di + d];
    const float Bm  = dbc[m * ldd + dr + s];
    const float Cm  = dbc[m * ldd + dr + 16 + s];
    h = __expf(dtv * A) * h + dtv * Bm * xv;
    float y = h * Cm;
    y += __shfl_xor(y, 8, 16);
    y += __shfl_xor(y, 4, 16);
    y += __shfl_xor(y, 2, 16);
    y += __shfl_xor(y, 1, 16);
    if (s == 0) {
      const float zv = zbase[m * ldz + d];
      out[m * di + d] = (y + xv * Dd) * siluf(zv);
    }
  }
}

// ---------------------------------------------------------------------------
// Elementwise / layout helpers
// ---------------------------------------------------------------------------
__global__ void k_act(const float* __restrict__ x, float* __restrict__ y, long n, int mode)
{
  const long i = (long)blockIdx.x * blockDim.x + threadIdx.x;
  if (i < n) y[i] = act_apply(x[i], mode);
}

__global__ void k_bn_act_mc(const float* __restrict__ x, const float* __restrict__ g,
                            const float* __restrict__ b, const float* __restrict__ rm,
                            const float* __restrict__ rv, float* __restrict__ y,
                            long M, int C, int act)
{
  const long i = (long)blockIdx.x * blockDim.x + threadIdx.x;
  if (i >= M * C) return;
  const int c = (int)(i % C);
  const float s = g[c] * rsqrtf(rv[c] + 1e-5f);
  y[i] = act_apply(x[i] * s + (b[c] - rm[c] * s), act);
}

__global__ void k_mean_h(const float* __restrict__ x, float* __restrict__ y,
                         int B, int C, int H, int W)   // y[b,c,w]
{
  const long i = (long)blockIdx.x * blockDim.x + threadIdx.x;
  if (i >= (long)B * C * W) return;
  const int w = (int)(i % W);
  const long t = i / W;
  const int c = (int)(t % C), b = (int)(t / C);
  float s = 0.f;
  for (int h = 0; h < H; ++h) s += x[(((size_t)b * C + c) * H + h) * W + w];
  y[i] = s / H;
}

__global__ void k_mean_w(const float* __restrict__ x, float* __restrict__ y,
                         int B, int C, int H, int W)   // y[b,c,h]
{
  const long i = (long)blockIdx.x * blockDim.x + threadIdx.x;
  if (i >= (long)B * C * H) return;
  const int h = (int)(i % H);
  const long t = i / H;
  const int c = (int)(t % C), b = (int)(t / C);
  float s = 0.f;
  const float* xr = &x[(((size_t)b * C + c) * H + h) * W];
  for (int w = 0; w < W; ++w) s += xr[w];
  y[i] = s / W;
}

__global__ void k_channel_att(const float* __restrict__ x, const float* __restrict__ fc1,
                              const float* __restrict__ fc2, float* __restrict__ a1,
                              int C, int HW, int Cm)
{
  __shared__ float g[128];
  __shared__ float hsh[8];
  const int b = blockIdx.x, c = threadIdx.x;
  float s = 0.f;
  const float* xb = x + ((size_t)b * C + c) * HW;
  for (int p = 0; p < HW; ++p) s += xb[p];
  g[c] = s / HW; __syncthreads();
  if (c < Cm) {
    float a = 0.f;
    for (int j = 0; j < C; ++j) a += fc1[c * C + j] * g[j];
    hsh[c] = fmaxf(a, 0.f);
  }
  __syncthreads();
  float a = 0.f;
  for (int j = 0; j < Cm; ++j) a += fc2[c * Cm + j] * hsh[j];
  a1[b * C + c] = sigm(a);
}

__global__ void k_hw_att(const float* __restrict__ xm, const float* __restrict__ c1w,
                         const float* __restrict__ c1b, const float* __restrict__ bng,
                         const float* __restrict__ bnb, const float* __restrict__ bnrm,
                         const float* __restrict__ bnrv, const float* __restrict__ c2w,
                         const float* __restrict__ c2b, float* __restrict__ a,
                         int C, int P, int Cm)
{
  __shared__ float g[128];
  __shared__ float t[8];
  const int b = blockIdx.x / P, p = blockIdx.x % P, c = threadIdx.x;
  g[c] = xm[((size_t)b * C + c) * P + p]; __syncthreads();
  if (c < Cm) {
    float v = c1b[c];
    for (int j = 0; j < C; ++j) v += c1w[c * C + j] * g[j];
    const float s = bng[c] * rsqrtf(bnrv[c] + 1e-5f);
    t[c] = hswishf(v * s + (bnb[c] - bnrm[c] * s));
  }
  __syncthreads();
  float v = c2b[c];
  for (int j = 0; j < Cm; ++j) v += c2w[c * Cm + j] * t[j];
  a[((size_t)b * C + c) * P + p] = sigm(v);
}

__global__ void k_apply_gates(const float* __restrict__ x, const float* __restrict__ a1,
                              const float* __restrict__ a2, const float* __restrict__ a3,
                              float* __restrict__ y, int B, int C, int H, int W)
{
  const long i = (long)blockIdx.x * blockDim.x + threadIdx.x;
  if (i >= (long)B * C * H * W) return;
  const int w = (int)(i % W);
  long t = i / W;
  const int h = (int)(t % H); t /= H;
  const int c = (int)(t % C), b = (int)(t / C);
  y[i] = x[i] * a1[b * C + c] * a2[((size_t)b * C + c) * W + w]
              * a3[((size_t)b * C + c) * H + h];
}

__global__ void k_copy_ch(const float* __restrict__ s, int Cs, int so,
                          float* __restrict__ d, int Cd, int dof,
                          int Cc, int B, int HW)
{
  const long i = (long)blockIdx.x * blockDim.x + threadIdx.x;
  if (i >= (long)B * Cc * HW) return;
  const int p = (int)(i % HW);
  const long t = i / HW;
  const int c = (int)(t % Cc), b = (int)(t / Cc);
  d[((size_t)b * Cd + dof + c) * HW + p] = s[((size_t)b * Cs + so + c) * HW + p];
}

__global__ void k_nchw2mc(const float* __restrict__ x, float* __restrict__ y,
                          int B, int C, int HW)
{
  const long i = (long)blockIdx.x * blockDim.x + threadIdx.x;
  if (i >= (long)B * C * HW) return;
  const int p = (int)(i % HW);
  const long t = i / HW;
  const int c = (int)(t % C), b = (int)(t / C);
  y[((size_t)b * HW + p) * C + c] = x[i];
}

__global__ void k_mc2nchw(const float* __restrict__ x, float* __restrict__ y,
                          int B, int C, int HW)
{
  const long i = (long)blockIdx.x * blockDim.x + threadIdx.x;
  if (i >= (long)B * C * HW) return;
  const int p = (int)(i % HW);
  const long t = i / HW;
  const int c = (int)(t % C), b = (int)(t / C);
  y[i] = x[((size_t)b * HW + p) * C + c];
}

__global__ void k_dwconv(const float* __restrict__ x, const float* __restrict__ w,
                         const float* __restrict__ bias, float* __restrict__ y,
                         int B, int C, int H, int W, int k, int pad)
{
  const long i = (long)blockIdx.x * blockDim.x + threadIdx.x;
  if (i >= (long)B * C * H * W) return;
  const int ww = (int)(i % W);
  long t = i / W;
  const int hh = (int)(t % H); t /= H;
  const int c = (int)(t % C), b = (int)(t / C);
  float acc = bias[c];
  for (int ky = 0; ky < k; ++ky) {
    const int ih = hh - pad + ky;
    if (ih < 0 || ih >= H) continue;
    for (int kx = 0; kx < k; ++kx) {
      const int iw = ww - pad + kx;
      if (iw < 0 || iw >= W) continue;
      acc += w[(c * k + ky) * k + kx] * x[(((size_t)b * C + c) * H + ih) * W + iw];
    }
  }
  y[i] = acc;
}

__global__ void k_amaxpool(const float* __restrict__ x, float* __restrict__ y,
                           int B, int C, int H, int W, int o)
{
  const long i = (long)blockIdx.x * blockDim.x + threadIdx.x;
  if (i >= (long)B * C * o * o) return;
  const int oj = (int)(i % o);
  long t = i / o;
  const int oi = (int)(t % o); t /= o;
  const int c = (int)(t % C), b = (int)(t / C);
  const int bh = H / o, bw = W / o;
  float m = -3.4e38f;
  for (int h = oi * bh; h < (oi + 1) * bh; ++h)
    for (int w = oj * bw; w < (oj + 1) * bw; ++w)
      m = fmaxf(m, x[(((size_t)b * C + c) * H + h) * W + w]);
  y[i] = m;
}

__global__ void k_bilinear(const float* __restrict__ x, float* __restrict__ y,
                           int B, int C, int h, int w, int H, int W)
{
  const long i = (long)blockIdx.x * blockDim.x + threadIdx.x;
  if (i >= (long)B * C * H * W) return;
  const int ox = (int)(i % W);
  long t = i / W;
  const int oy = (int)(t % H); t /= H;
  const int c = (int)(t % C), b = (int)(t / C);
  float fy = fmaxf((oy + 0.5f) * h / H - 0.5f, 0.f);
  float fx = fmaxf((ox + 0.5f) * w / W - 0.5f, 0.f);
  const int y0 = min((int)fy, h - 1), x0 = min((int)fx, w - 1);
  const int y1 = min(y0 + 1, h - 1), x1 = min(x0 + 1, w - 1);
  const float wy = fy - y0, wx = fx - x0;
  const float* xb = x + ((size_t)b * C + c) * h * w;
  const float v = (1.f - wy) * ((1.f - wx) * xb[y0 * w + x0] + wx * xb[y0 * w + x1])
                +        wy  * ((1.f - wx) * xb[y1 * w + x0] + wx * xb[y1 * w + x1]);
  y[i] = v;
}

// ===========================================================================
// Host orchestration
// ===========================================================================
namespace {

constexpr int Bb = 4, DIM = 128, Hh = 60, Ww = 60, HW = 3600, Mtok = Bb * HW;

static inline int cdiv(long a, long b) { return (int)((a + b - 1) / b); }

struct Bump {
  char* p;
  float* take(size_t nfloats) {
    float* r = (float*)p;
    p += (nfloats * sizeof(float) + 255) & ~(size_t)255;
    return r;
  }
};

static void gemm(hipStream_t st, const float* A, const float* W, const float* bias,
                 const float* pm, float* C, int M, int N, int K, int act)
{
  dim3 g(cdiv(N, BN), cdiv(M, BM));
  k_gemm_wmma<<<g, 256, 0, st>>>(A, W, bias, pm, C, M, N, K, act);
}

#define EW(n) <<<cdiv((long)(n), 256), 256, 0, st>>>

// Param leaf indices (alphabetical recursive flatten; d_in[0] = x).
enum {
  IN_X = 0,
  AH_BN_B, AH_BN_G, AH_BN_RM, AH_BN_RV, AH_C1_B, AH_C1_W, AH_C2_B, AH_C2_W,
  AW_BN_B, AW_BN_G, AW_BN_RM, AW_BN_RV, AW_C1_B, AW_C1_W, AW_C2_B, AW_C2_W,
  B1_DB, B1_DW,
  B1M_BASE,                 // 11 mamba leaves: A_log,D,conv_b,conv_w,dt_b,dt_w,in_w,ln_b,ln_w,out_w,xp_w
  CA_FC1 = B1M_BASE + 11, CA_FC2,
  CBN_B, CBN_G, CBN_RM, CBN_RV,
  CONV_W,
  MAB_BASE,
  PEBN_B = MAB_BASE + 11, PEBN_G, PEBN_RM, PEBN_RV,
  PEND_W,
  FFN_DWB, FFN_DWW, FFN_GAMMA, FFN_LNB, FFN_LNW, FFN_PW1B, FFN_PW1W, FFN_PW2B, FFN_PW2W,
  PMAB_BASE,
  PS2_C1B = PMAB_BASE + 11, PS2_C1W, PS2_C2B, PS2_C2W,
  PS2M_BASE,
  PS3_C1B = PS2M_BASE + 11, PS3_C1W, PS3_C2B, PS3_C2W,
  PS3M_BASE,
  N_IN = PS3M_BASE + 11     // == 103
};

struct MambaP {
  const float *A_log, *D, *conv_b, *conv_w, *dt_b, *dt_w, *in_w, *ln_b, *ln_w, *out_w, *xp_w;
  int dm, di, dr;
};

static MambaP mk_mamba(void* const* d_in, int base, int dm) {
  auto P = [&](int i) { return (const float*)d_in[i]; };
  MambaP m;
  m.A_log = P(base + 0);  m.D     = P(base + 1); m.conv_b = P(base + 2);
  m.conv_w= P(base + 3);  m.dt_b  = P(base + 4); m.dt_w   = P(base + 5);
  m.in_w  = P(base + 6);  m.ln_b  = P(base + 7); m.ln_w   = P(base + 8);
  m.out_w = P(base + 9);  m.xp_w  = P(base + 10);
  m.dm = dm; m.di = 2 * dm; m.dr = (dm + 15) / 16;
  return m;
}

// mamba() on token-major [B*L, dm] input -> [B*L, dm] output (includes the
// preceding LayerNorm of mamba_layer)
static void run_mamba(hipStream_t st, const float* x_mc, float* out_mc,
                      const MambaP& p, int B, int L, Bump& ws)
{
  const int M = B * L, dm = p.dm, di = p.di, dr = p.dr, ldd = dr + 32;
  float* lnb = ws.take((size_t)M * dm);
  float* xz  = ws.take((size_t)M * 2 * di);
  float* xs  = ws.take((size_t)M * di);
  float* dbc = ws.take((size_t)M * ldd);
  float* dtb = ws.take((size_t)M * di);
  float* yg  = ws.take((size_t)M * di);

  k_ln<<<M, 128, 0, st>>>(x_mc, p.ln_w, p.ln_b, lnb, M, dm, 1e-5f);
  gemm(st, lnb, p.in_w, nullptr, nullptr, xz, M, 2 * di, dm, 0);
  k_conv1d_silu EW((long)M * di)(xz, 2 * di, 0, p.conv_w, p.conv_b, xs, B, L, di);
  gemm(st, xs, p.xp_w, nullptr, nullptr, dbc, M, ldd, di, 0);
  k_dt_softplus EW((long)M * di)(dbc, ldd, p.dt_w, p.dt_b, dtb, M, di, dr);
  k_scan<<<B * (di / 16), 256, 0, st>>>(dtb, xs, dbc, ldd, dr, xz + di, 2 * di,
                                        p.A_log, p.D, yg, B, L, di);
  gemm(st, yg, p.out_w, nullptr, nullptr, out_mc, M, dm, di, 0);
}

// PPM pooled-mamba branch: pool(o) -> 1x1 -> mamba(dm=8) -> gelu -> 1x1 -> resize
static void run_ps_branch(hipStream_t st, void* const* d_in, const float* xa_nchw,
                          float* out_nchw, int o, int c1b, int c1w, int c2b, int c2w,
                          int mbase, Bump& ws)
{
  auto P = [&](int i) { return (const float*)d_in[i]; };
  const int P2 = o * o, M = Bb * P2;
  float* pool = ws.take((size_t)Bb * 64 * P2);
  float* mc64 = ws.take((size_t)M * 64);
  float* c1o  = ws.take((size_t)M * 8);
  float* mo   = ws.take((size_t)M * 8);
  float* c2o  = ws.take((size_t)M * 64);
  float* c2n  = ws.take((size_t)Bb * 64 * P2);

  k_amaxpool EW((long)Bb * 64 * P2)(xa_nchw, pool, Bb, 64, Hh, Ww, o);
  k_nchw2mc EW((long)Bb * 64 * P2)(pool, mc64, Bb, 64, P2);
  gemm(st, mc64, P(c1w), P(c1b), nullptr, c1o, M, 8, 64, 0);
  MambaP m = mk_mamba(d_in, mbase, 8);
  run_mamba(st, c1o, mo, m, Bb, P2, ws);
  k_act EW((long)M * 8)(mo, mo, (long)M * 8, 2 /*gelu*/);
  gemm(st, mo, P(c2w), P(c2b), nullptr, c2o, M, 64, 8, 0);
  k_mc2nchw EW((long)Bb * 64 * P2)(c2o, c2n, Bb, 64, P2);
  k_bilinear EW((long)Bb * 64 * HW)(c2n, out_nchw, Bb, 64, o, o, Hh, Ww);
}

} // namespace

extern "C" void kernel_launch(void* const* d_in, const int* in_sizes, int n_in,
                              void* d_out, int out_size, void* d_ws, size_t ws_size,
                              hipStream_t stream)
{
  (void)in_sizes; (void)n_in; (void)out_size; (void)ws_size;
  hipStream_t st = stream;
  auto P = [&](int i) { return (const float*)d_in[i]; };
  Bump ws{(char*)d_ws};

  const float* x = P(IN_X);

  // ---- attention gates -------------------------------------------------
  float* a1 = ws.take(Bb * DIM);
  float* xw = ws.take(Bb * DIM * Ww);
  float* xh = ws.take(Bb * DIM * Hh);
  float* a2 = ws.take(Bb * DIM * Ww);
  float* a3 = ws.take(Bb * DIM * Hh);
  float* xg = ws.take((size_t)Bb * DIM * HW);

  k_channel_att<<<Bb, DIM, 0, st>>>(x, P(CA_FC1), P(CA_FC2), a1, DIM, HW, 8);
  k_mean_h EW((long)Bb * DIM * Ww)(x, xw, Bb, DIM, Hh, Ww);
  k_mean_w EW((long)Bb * DIM * Hh)(x, xh, Bb, DIM, Hh, Ww);
  k_hw_att<<<Bb * Ww, DIM, 0, st>>>(xw, P(AW_C1_W), P(AW_C1_B), P(AW_BN_G), P(AW_BN_B),
                                    P(AW_BN_RM), P(AW_BN_RV), P(AW_C2_W), P(AW_C2_B),
                                    a2, DIM, Ww, 8);
  k_hw_att<<<Bb * Hh, DIM, 0, st>>>(xh, P(AH_C1_W), P(AH_C1_B), P(AH_BN_G), P(AH_BN_B),
                                    P(AH_BN_RM), P(AH_BN_RV), P(AH_C2_W), P(AH_C2_B),
                                    a3, DIM, Hh, 8);
  k_apply_gates EW((long)Bb * DIM * HW)(x, a1, a2, a3, xg, Bb, DIM, Hh, Ww);

  // ---- split -----------------------------------------------------------
  float* xa = ws.take((size_t)Bb * 64 * HW);
  float* xb = ws.take((size_t)Bb * 64 * HW);
  k_copy_ch EW((long)Bb * 64 * HW)(xg, DIM, 0, xa, 64, 0, 64, Bb, HW);
  k_copy_ch EW((long)Bb * 64 * HW)(xg, DIM, 64, xb, 64, 0, 64, Bb, HW);

  // ---- PPM: dwconv-FFN -------------------------------------------------
  float* dw7  = ws.take((size_t)Bb * 64 * HW);
  float* fmc  = ws.take((size_t)Mtok * 64);
  float* flnb = ws.take((size_t)Mtok * 64);
  float* p1   = ws.take((size_t)Mtok * 256);
  float* p2   = ws.take((size_t)Mtok * 64);
  float* x1a  = ws.take((size_t)Bb * 64 * HW);

  k_dwconv EW((long)Bb * 64 * HW)(xa, P(FFN_DWW), P(FFN_DWB), dw7, Bb, 64, Hh, Ww, 7, 3);
  k_nchw2mc EW((long)Bb * 64 * HW)(dw7, fmc, Bb, 64, HW);
  k_ln<<<Mtok, 128, 0, st>>>(fmc, P(FFN_LNW), P(FFN_LNB), flnb, Mtok, 64, 1e-6f);
  gemm(st, flnb, P(FFN_PW1W), P(FFN_PW1B), nullptr, p1, Mtok, 256, 64, 2 /*gelu*/);
  gemm(st, p1, P(FFN_PW2W), P(FFN_PW2B), P(FFN_GAMMA), p2, Mtok, 64, 256, 0);
  k_mc2nchw EW((long)Bb * 64 * HW)(p2, x1a, Bb, 64, HW);

  // ---- PPM: pooled mamba branches --------------------------------------
  float* r2 = ws.take((size_t)Bb * 64 * HW);   // ps2, o=3
  float* r3 = ws.take((size_t)Bb * 64 * HW);   // ps3, o=5
  run_ps_branch(st, d_in, xa, r2, 3, PS2_C1B, PS2_C1W, PS2_C2B, PS2_C2W, PS2M_BASE, ws);
  run_ps_branch(st, d_in, xa, r3, 5, PS3_C1B, PS3_C1W, PS3_C2B, PS3_C2W, PS3M_BASE, ws);

  // ---- PPM: concat([xa, x3, x2, x1a]) -> end conv -> bn relu -> mamba64 --
  float* cat4  = ws.take((size_t)Bb * 256 * HW);
  float* cat4m = ws.take((size_t)Mtok * 256);
  float* endo  = ws.take((size_t)Mtok * 64);
  float* x1mc  = ws.take((size_t)Mtok * 64);
  float* x1n   = ws.take((size_t)Bb * 64 * HW);

  k_copy_ch EW((long)Bb * 64 * HW)(xa,  64, 0, cat4, 256,   0, 64, Bb, HW);
  k_copy_ch EW((long)Bb * 64 * HW)(r3,  64, 0, cat4, 256,  64, 64, Bb, HW);
  k_copy_ch EW((long)Bb * 64 * HW)(r2,  64, 0, cat4, 256, 128, 64, Bb, HW);
  k_copy_ch EW((long)Bb * 64 * HW)(x1a, 64, 0, cat4, 256, 192, 64, Bb, HW);
  k_nchw2mc EW((long)Bb * 256 * HW)(cat4, cat4m, Bb, 256, HW);
  gemm(st, cat4m, P(PEND_W), nullptr, nullptr, endo, Mtok, 64, 256, 0);
  k_bn_act_mc EW((long)Mtok * 64)(endo, P(PEBN_G), P(PEBN_B), P(PEBN_RM), P(PEBN_RV),
                                  endo, Mtok, 64, 1 /*relu*/);
  {
    MambaP m = mk_mamba(d_in, PMAB_BASE, 64);
    run_mamba(st, endo, x1mc, m, Bb, HW, ws);
  }
  k_mc2nchw EW((long)Bb * 64 * HW)(x1mc, x1n, Bb, 64, HW);

  // ---- parallel branch: dw3x3 -> mamba64 -> relu -----------------------
  float* dw3  = ws.take((size_t)Bb * 64 * HW);
  float* bmc  = ws.take((size_t)Mtok * 64);
  float* x2mc = ws.take((size_t)Mtok * 64);
  float* x2n  = ws.take((size_t)Bb * 64 * HW);
  k_dwconv EW((long)Bb * 64 * HW)(xb, P(B1_DW), P(B1_DB), dw3, Bb, 64, Hh, Ww, 3, 1);
  k_nchw2mc EW((long)Bb * 64 * HW)(dw3, bmc, Bb, 64, HW);
  {
    MambaP m = mk_mamba(d_in, B1M_BASE, 64);
    run_mamba(st, bmc, x2mc, m, Bb, HW, ws);
  }
  k_act EW((long)Mtok * 64)(x2mc, x2mc, (long)Mtok * 64, 1 /*relu*/);
  k_mc2nchw EW((long)Bb * 64 * HW)(x2mc, x2n, Bb, 64, HW);

  // ---- fuse: concat([xg, x1, x2]) -> 1x1 conv -> bn relu -> mamba128 ---
  float* caty  = ws.take((size_t)Bb * 256 * HW);
  float* catym = ws.take((size_t)Mtok * 256);
  float* convo = ws.take((size_t)Mtok * DIM);
  float* ymc   = ws.take((size_t)Mtok * DIM);

  k_copy_ch EW((long)Bb * DIM * HW)(xg, DIM, 0, caty, 256,   0, DIM, Bb, HW);
  k_copy_ch EW((long)Bb * 64 * HW)(x1n, 64, 0, caty, 256, 128, 64, Bb, HW);
  k_copy_ch EW((long)Bb * 64 * HW)(x2n, 64, 0, caty, 256, 192, 64, Bb, HW);
  k_nchw2mc EW((long)Bb * 256 * HW)(caty, catym, Bb, 256, HW);
  gemm(st, catym, P(CONV_W), nullptr, nullptr, convo, Mtok, DIM, 256, 0);
  k_bn_act_mc EW((long)Mtok * DIM)(convo, P(CBN_G), P(CBN_B), P(CBN_RM), P(CBN_RV),
                                   convo, Mtok, DIM, 1 /*relu*/);
  {
    MambaP m = mk_mamba(d_in, MAB_BASE, 128);
    run_mamba(st, convo, ymc, m, Bb, HW, ws);
  }
  k_mc2nchw EW((long)Bb * DIM * HW)(ymc, (float*)d_out, Bb, DIM, HW);
}
